// RelationAwareGraphStateEncoder_51823075393575
// MI455X (gfx1250) — matrix operand order repaired and checked by reading, not compiled, not run on hardware
//
#include <hip/hip_runtime.h>
#include <hip/hip_bf16.h>

typedef __attribute__((ext_vector_type(16))) __bf16 v16bf;
typedef __attribute__((ext_vector_type(8)))  __bf16 v8bf;
typedef __attribute__((ext_vector_type(8)))  float  v8f;

#define HDIM 128
#define LDS_STRIDE 144   // 16-elem pad: keeps 16B alignment, staggers banks

// ---------------- WMMA fragment loaders ----------------
// A (16x32 bf16, row-major source):
// lane m (0..15), hi = lane>>4 : halves 0..7 = K[k0+hi*8 .. +7],
//                                halves 8..15 = K[k0+16+hi*8 .. +7]
__device__ __forceinline__ v16bf load_fragA(const __bf16* row, int k0, int hi) {
  v8bf x = *(const v8bf*)(row + k0 + hi * 8);
  v8bf y = *(const v8bf*)(row + k0 + 16 + hi * 8);
  return __builtin_shufflevector(x, y, 0,1,2,3,4,5,6,7,8,9,10,11,12,13,14,15);
}

// B (32x16 bf16) from transposed weights Wt[outCol][k] (k-stride 128):
// lane n (0..15), kh = lane>>4 : halves 0..7 = K[k0+kh*8 .. +7], halves 8..15 = K[k0+16+kh*8 .. +7]
__device__ __forceinline__ v16bf load_fragB(const __bf16* colrow, int k0, int kh) {
  const __bf16* p = colrow + k0 + kh * 8;
  v8bf x = *(const v8bf*)p;
  v8bf y = *(const v8bf*)(p + 16);
  return __builtin_shufflevector(x, y, 0,1,2,3,4,5,6,7,8,9,10,11,12,13,14,15);
}

__device__ __forceinline__ v8f wmma_bf16(v16bf a, v16bf b, v8f c) {
  return __builtin_amdgcn_wmma_f32_16x16x32_bf16(false, a, false, b, (short)0, c,
                                                 false, false);
}

// ---- async global -> LDS staging of one 32KB weight chunk (128 threads) ----
__device__ __forceinline__ void stage_issue(__bf16* ldsbuf, const __bf16* gsrc,
                                            int tid) {
  const char* g = (const char*)gsrc;
  char* l = (char*)ldsbuf;
#pragma unroll
  for (int i = 0; i < 16; ++i) {
    int off = (tid + i * 128) * 16;  // 128 thr * 16B * 16 = 32768B
    asm volatile("global_load_async_to_lds_b128 %0, %1, off"
                 :: "v"((unsigned)(uintptr_t)(l + off)), "v"(g + off)
                 : "memory");
  }
}
__device__ __forceinline__ void async_wait0() {
  asm volatile("s_wait_asynccnt 0x0" ::: "memory");
}

// one 16x128 @ 128x128 accumulate step (4 K-tiles), B staged in LDS
__device__ __forceinline__ void gemm_acc(v8f acc[8], const __bf16* rowA,
                                         const __bf16* wsw, int m, int hi) {
#pragma unroll
  for (int kt = 0; kt < 4; ++kt) {
    v16bf a = load_fragA(rowA, kt * 32, hi);
    v16bf b[8];
#pragma unroll
    for (int nt = 0; nt < 8; ++nt)
      b[nt] = load_fragB(wsw + (nt * 16 + m) * 128, kt * 32, hi);
#pragma unroll
    for (int nt = 0; nt < 8; ++nt) acc[nt] = wmma_bf16(a, b[nt], acc[nt]);
  }
}

// ---------------- prep kernels ----------------
__global__ __launch_bounds__(256) void k_zero(float* agg, float* sc, int n) {
  int i = blockIdx.x * 256 + threadIdx.x;
  if (i < n) agg[i] = 0.0f;
  if (i == 0) { sc[0] = 1.0f; sc[1] = 1.0f; sc[2] = 0.0f; }
}

__global__ __launch_bounds__(256) void k_reduce(const float* bw, const float* pw,
                                                const float* area, float* sc,
                                                int E1, int E2, int N) {
  __shared__ float r0[256], r1[256], r2[256];
  int tid = threadIdx.x;
  int gid = blockIdx.x * 256 + tid;
  int stride = gridDim.x * 256;
  float m0 = 0.f, m1 = 0.f, s = 0.f;
  for (int i = gid; i < E1; i += stride) m0 = fmaxf(m0, fabsf(bw[i]));
  for (int i = gid; i < E2; i += stride) m1 = fmaxf(m1, fabsf(pw[i]));
  for (int i = gid; i < N; i += stride) s += area[i];
  r0[tid] = m0; r1[tid] = m1; r2[tid] = s;
  __syncthreads();
  for (int k = 128; k > 0; k >>= 1) {
    if (tid < k) {
      r0[tid] = fmaxf(r0[tid], r0[tid + k]);
      r1[tid] = fmaxf(r1[tid], r1[tid + k]);
      r2[tid] += r2[tid + k];
    }
    __syncthreads();
  }
  if (tid == 0) {
    atomicMax((int*)&sc[0], __float_as_int(r0[0]));
    atomicMax((int*)&sc[1], __float_as_int(r1[0]));
    atomicAdd(&sc[2], r2[0]);
  }
}

// transpose + f32->bf16 convert of edge-MLP weights into [out][k] chunks
__global__ __launch_bounds__(256) void k_prep(const float* b2bW1, const float* b2bW2,
                                              const float* p2bW1, const float* p2bW2,
                                              __bf16* t1a, __bf16* t1b, __bf16* t2,
                                              __bf16* t3, __bf16* t4) {
  int i = blockIdx.x * 256 + threadIdx.x;
  if (i < 128 * 128) {
    int n = i >> 7, k = i & 127;
    t1a[n * 128 + k] = (__bf16)b2bW1[k * HDIM + n];
    t1b[n * 128 + k] = (__bf16)b2bW1[(k + 128) * HDIM + n];
    t2[n * 128 + k] = (__bf16)b2bW2[k * HDIM + n];
    t3[n * 128 + k] = (__bf16)p2bW1[k * HDIM + n];
    t4[n * 128 + k] = (__bf16)p2bW2[k * HDIM + n];
  }
}

// ---------------- input projection: h = relu(relu(x@W1+b1)@W2+b2) ----------------
__global__ __launch_bounds__(128) void k_input(const float* bf, const int* roles,
                                               const float* role_emb,
                                               const float* idx_emb, const float* W1,
                                               const float* B1, const float* W2,
                                               const float* B2, float* hf, __bf16* hb) {
  __shared__ float x[40];
  __shared__ float t[128];
  int n = blockIdx.x, j = threadIdx.x;
  if (j < 16) x[j] = bf[n * 16 + j];
  else if (j < 32) x[j] = role_emb[roles[n] * 16 + (j - 16)];
  else if (j < 40) x[j] = idx_emb[(n & 1023) * 8 + (j - 32)];
  __syncthreads();
  float s = B1[j];
#pragma unroll 8
  for (int k = 0; k < 40; ++k) s = fmaf(x[k], W1[k * HDIM + j], s);
  t[j] = s > 0.f ? s : 0.f;
  __syncthreads();
  float s2 = B2[j];
#pragma unroll 8
  for (int k = 0; k < 128; ++k) s2 = fmaf(t[k], W2[k * HDIM + j], s2);
  s2 = s2 > 0.f ? s2 : 0.f;
  hf[n * HDIM + j] = s2;
  hb[n * HDIM + j] = (__bf16)s2;
}

// -------- b2b edge messages: 4 waves/block, one 16-edge tile per wave --------
__global__ __launch_bounds__(128) void k_b2b(const int* src, const int* dst,
                                             const float* wE, const __bf16* hb,
                                             const __bf16* w1a, const __bf16* w1b,
                                             const __bf16* w2t, const float* W1full,
                                             const float* b1, const float* b2,
                                             const float* sc, float* agg, int E1) {
  __shared__ __bf16 wsw[128 * 128];             // 32KB staged weight chunk
  __shared__ __bf16 su[4][16 * LDS_STRIDE];     // per-wave activation re-layout
  __shared__ int sa[4][16];
  __shared__ int sb[4][16];
  __shared__ float sw[4][16];
  int tid = threadIdx.x;
  int lane = tid & 31, wid = tid >> 5;
  int m = lane & 15, hi = lane >> 4;
  int e0 = (blockIdx.x * 4 + wid) * 16;
  float invmax = 1.0f / sc[0];
  if (lane < 16) {
    int e = e0 + lane;
    int a, b; float w;
    if (e < E1) { a = src[e]; b = dst[e]; w = wE[e]; }
    else        { a = dst[e - E1]; b = src[e - E1]; w = wE[e - E1]; }
    sa[wid][lane] = a; sb[wid][lane] = b; sw[wid][lane] = w * invmax;
  }
  // stage W1[0:128,:]^T while metadata lands
  stage_issue(wsw, w1a, tid);
  async_wait0();
  __syncthreads();

  const __bf16* rowA = hb + (size_t)sa[wid][m] * HDIM;
  const __bf16* rowB = hb + (size_t)sb[wid][m] * HDIM;

  v8f acc[8] = {};
  gemm_acc(acc, rowA, wsw, m, hi);       // hi @ W1[0:128]
  __syncthreads();
  stage_issue(wsw, w1b, tid);            // W1[128:256,:]^T
  async_wait0();
  __syncthreads();
  gemm_acc(acc, rowB, wsw, m, hi);       // + hj @ W1[128:256]
  __syncthreads();
  stage_issue(wsw, w2t, tid);            // issue W2^T; overlap with epilogue VALU

  // epilogue: + w*W1[256,:] + b1, ReLU, stash as bf16 A-operand in LDS
  int mb = hi * 8;
#pragma unroll
  for (int nt = 0; nt < 8; ++nt) {
    int gn = nt * 16 + m;
    float wr = W1full[256 * HDIM + gn];
    float bb = b1[gn];
    v8f c = acc[nt];
#pragma unroll
    for (int r = 0; r < 8; ++r) {
      float v = c[r] + sw[wid][r + mb] * wr + bb;
      v = v > 0.f ? v : 0.f;
      su[wid][(r + mb) * LDS_STRIDE + gn] = (__bf16)v;
    }
  }
  async_wait0();
  __syncthreads();

  // GEMM2: msg = u @ W2 + b2
  v8f acc2[8] = {};
#pragma unroll
  for (int kt = 0; kt < 4; ++kt) {
    v16bf a = load_fragA(&su[wid][m * LDS_STRIDE], kt * 32, hi);
    v16bf b[8];
#pragma unroll
    for (int nt = 0; nt < 8; ++nt)
      b[nt] = load_fragB(wsw + (nt * 16 + m) * 128, kt * 32, hi);
#pragma unroll
    for (int nt = 0; nt < 8; ++nt) acc2[nt] = wmma_bf16(a, b[nt], acc2[nt]);
  }
  // scatter-add to agg[a_node]
#pragma unroll
  for (int nt = 0; nt < 8; ++nt) {
    int gn = nt * 16 + m;
    float bb = b2[gn];
    v8f c = acc2[nt];
#pragma unroll
    for (int r = 0; r < 8; ++r) {
      int node = sa[wid][r + mb];
      atomicAdd(&agg[(size_t)node * HDIM + gn], c[r] + bb);
    }
  }
}

// -------- p2b edge messages: 4 waves/block, one 16-edge tile per wave --------
__global__ __launch_bounds__(128) void k_p2b(const int* pin, const int* blk,
                                             const float* wE, const float* pins_pos,
                                             const __bf16* hb, const __bf16* w1t,
                                             const __bf16* w2t, const float* W1full,
                                             const float* b1, const float* b2,
                                             const float* sc, float* agg) {
  __shared__ __bf16 wsw[128 * 128];
  __shared__ __bf16 su[4][16 * LDS_STRIDE];
  __shared__ int snode[4][16];
  __shared__ float spx[4][16], spy[4][16], spw[4][16];
  int tid = threadIdx.x;
  int lane = tid & 31, wid = tid >> 5;
  int m = lane & 15, hi = lane >> 4;
  int e0 = (blockIdx.x * 4 + wid) * 16;
  float invmax = 1.0f / sc[1];
  float ta = fmaxf(sc[2], 1e-6f);
  float inv_cs = 1.0f / fmaxf(sqrtf(ta), 1e-6f);
  if (lane < 16) {
    int e = e0 + lane;
    int p = pin[e];
    snode[wid][lane] = blk[e];
    spx[wid][lane] = pins_pos[p * 2 + 0] * inv_cs;
    spy[wid][lane] = pins_pos[p * 2 + 1] * inv_cs;
    spw[wid][lane] = wE[e] * invmax;
  }
  stage_issue(wsw, w1t, tid);
  async_wait0();
  __syncthreads();

  const __bf16* rowA = hb + (size_t)snode[wid][m] * HDIM;
  v8f acc[8] = {};
  gemm_acc(acc, rowA, wsw, m, hi);
  __syncthreads();
  stage_issue(wsw, w2t, tid);  // overlap with epilogue

  int mb = hi * 8;
#pragma unroll
  for (int nt = 0; nt < 8; ++nt) {
    int gn = nt * 16 + m;
    float r0 = W1full[128 * HDIM + gn];
    float r1 = W1full[129 * HDIM + gn];
    float r2 = W1full[130 * HDIM + gn];
    float bb = b1[gn];
    v8f c = acc[nt];
#pragma unroll
    for (int r = 0; r < 8; ++r) {
      int rm = r + mb;
      float v = c[r] + spx[wid][rm] * r0 + spy[wid][rm] * r1 + spw[wid][rm] * r2 + bb;
      v = v > 0.f ? v : 0.f;
      su[wid][rm * LDS_STRIDE + gn] = (__bf16)v;
    }
  }
  async_wait0();
  __syncthreads();

  v8f acc2[8] = {};
#pragma unroll
  for (int kt = 0; kt < 4; ++kt) {
    v16bf a = load_fragA(&su[wid][m * LDS_STRIDE], kt * 32, hi);
    v16bf b[8];
#pragma unroll
    for (int nt = 0; nt < 8; ++nt)
      b[nt] = load_fragB(wsw + (nt * 16 + m) * 128, kt * 32, hi);
#pragma unroll
    for (int nt = 0; nt < 8; ++nt) acc2[nt] = wmma_bf16(a, b[nt], acc2[nt]);
  }
#pragma unroll
  for (int nt = 0; nt < 8; ++nt) {
    int gn = nt * 16 + m;
    float bb = b2[gn];
    v8f c = acc2[nt];
#pragma unroll
    for (int r = 0; r < 8; ++r) {
      int node = snode[wid][r + mb];
      atomicAdd(&agg[(size_t)node * HDIM + gn], c[r] + bb);
    }
  }
}

// ---------------- combine + LayerNorm ----------------
__global__ __launch_bounds__(128) void k_combine(const float* hf, const float* agg,
                                                 const float* sW, const float* sB,
                                                 const float* g, const float* be,
                                                 float* out) {
  __shared__ float hrow[128];
  __shared__ float red[128];
  int n = blockIdx.x, j = threadIdx.x;
  hrow[j] = hf[(size_t)n * HDIM + j];
  __syncthreads();
  float z = sB[j] + agg[(size_t)n * HDIM + j];
#pragma unroll 8
  for (int k = 0; k < 128; ++k) z = fmaf(hrow[k], sW[k * HDIM + j], z);
  red[j] = z;
  __syncthreads();
  for (int k = 64; k > 0; k >>= 1) {
    if (j < k) red[j] += red[j + k];
    __syncthreads();
  }
  float mu = red[0] * (1.0f / 128.0f);
  __syncthreads();
  float d = z - mu;
  red[j] = d * d;
  __syncthreads();
  for (int k = 64; k > 0; k >>= 1) {
    if (j < k) red[j] += red[j + k];
    __syncthreads();
  }
  float var = red[0] * (1.0f / 128.0f);
  out[(size_t)n * HDIM + j] = d * rsqrtf(var + 1e-5f) * g[j] + be[j];
}

// ---------------- pooling + graph MLP ----------------
__global__ __launch_bounds__(128) void k_graph(const float* BE, const float* W1,
                                               const float* B1, const float* W2,
                                               const float* B2, float* out, int N) {
  __shared__ float pooled[256];
  __shared__ float hid[128];
  int j = threadIdx.x;
  float s = 0.f, mx = -3.4e38f;
  for (int n = 0; n < N; ++n) {
    float v = BE[(size_t)n * HDIM + j];
    s += v;
    mx = fmaxf(mx, v);
  }
  pooled[j] = s / (float)N;
  pooled[128 + j] = mx;
  __syncthreads();
  float t = B1[j];
#pragma unroll 8
  for (int k = 0; k < 256; ++k) t = fmaf(pooled[k], W1[k * HDIM + j], t);
  hid[j] = t > 0.f ? t : 0.f;
  __syncthreads();
  float o = B2[j];
#pragma unroll 8
  for (int k = 0; k < 128; ++k) o = fmaf(hid[k], W2[k * HDIM + j], o);
  out[j] = o;
}

__global__ __launch_bounds__(256) void k_mask(float* out, int N) {
  int i = blockIdx.x * 256 + threadIdx.x;
  if (i < N) out[i] = 1.0f;
}

// ---------------- host launch ----------------
extern "C" void kernel_launch(void* const* d_in, const int* in_sizes, int n_in,
                              void* d_out, int out_size, void* d_ws, size_t ws_size,
                              hipStream_t stream) {
  const float* block_features = (const float*)d_in[0];
  const int*   role_ids       = (const int*)d_in[1];
  const int*   b2b_src        = (const int*)d_in[2];
  const int*   b2b_dst        = (const int*)d_in[3];
  const float* b2b_w          = (const float*)d_in[4];
  const int*   p2b_pin        = (const int*)d_in[5];
  const int*   p2b_block      = (const int*)d_in[6];
  const float* p2b_w          = (const float*)d_in[7];
  const float* pins_pos       = (const float*)d_in[8];
  const float* area_targets   = (const float*)d_in[9];
  const float* role_emb       = (const float*)d_in[10];
  const float* idx_emb        = (const float*)d_in[11];
  const float* in_W1 = (const float*)d_in[12];
  const float* in_b1 = (const float*)d_in[13];
  const float* in_W2 = (const float*)d_in[14];
  const float* in_b2 = (const float*)d_in[15];
  const float* b2b_W1 = (const float*)d_in[16];
  const float* b2b_b1 = (const float*)d_in[17];
  const float* b2b_W2 = (const float*)d_in[18];
  const float* b2b_b2 = (const float*)d_in[19];
  const float* p2b_W1 = (const float*)d_in[20];
  const float* p2b_b1 = (const float*)d_in[21];
  const float* p2b_W2 = (const float*)d_in[22];
  const float* p2b_b2 = (const float*)d_in[23];
  const float* self_W = (const float*)d_in[24];
  const float* self_b = (const float*)d_in[25];
  const float* ln_g = (const float*)d_in[26];
  const float* ln_b = (const float*)d_in[27];
  const float* gp_W1 = (const float*)d_in[28];
  const float* gp_b1 = (const float*)d_in[29];
  const float* gp_W2 = (const float*)d_in[30];
  const float* gp_b2 = (const float*)d_in[31];

  int N  = in_sizes[1];
  int E1 = in_sizes[2];
  int E2 = in_sizes[5];

  char* w = (char*)d_ws;
  float* sc = (float*)w;            w += 64;
  float* agg = (float*)w;           w += (size_t)N * HDIM * 4;
  float* hf = (float*)w;            w += (size_t)N * HDIM * 4;
  __bf16* hb = (__bf16*)w;          w += (size_t)N * HDIM * 2;
  __bf16* t1a = (__bf16*)w;         w += 128 * 128 * 2;  // b2b_W1[0:128,:]^T
  __bf16* t1b = (__bf16*)w;         w += 128 * 128 * 2;  // b2b_W1[128:256,:]^T
  __bf16* t2 = (__bf16*)w;          w += 128 * 128 * 2;  // b2b_W2^T
  __bf16* t3 = (__bf16*)w;          w += 128 * 128 * 2;  // p2b_W1[0:128,:]^T
  __bf16* t4 = (__bf16*)w;          w += 128 * 128 * 2;  // p2b_W2^T

  float* out = (float*)d_out;

  k_zero<<<(N * HDIM + 255) / 256, 256, 0, stream>>>(agg, sc, N * HDIM);
  k_reduce<<<256, 256, 0, stream>>>(b2b_w, p2b_w, area_targets, sc, E1, E2, N);
  k_prep<<<64, 256, 0, stream>>>(b2b_W1, b2b_W2, p2b_W1, p2b_W2, t1a, t1b, t2, t3, t4);
  k_input<<<N, 128, 0, stream>>>(block_features, role_ids, role_emb, idx_emb, in_W1,
                                 in_b1, in_W2, in_b2, hf, hb);
  k_b2b<<<(2 * E1) / 64, 128, 0, stream>>>(b2b_src, b2b_dst, b2b_w, hb, t1a, t1b, t2,
                                           b2b_W1, b2b_b1, b2b_b2, sc, agg, E1);
  k_p2b<<<E2 / 64, 128, 0, stream>>>(p2b_pin, p2b_block, p2b_w, pins_pos, hb, t3, t4,
                                     p2b_W1, p2b_b1, p2b_b2, sc, agg);
  k_combine<<<N, 128, 0, stream>>>(hf, agg, self_W, self_b, ln_g, ln_b, out);
  k_graph<<<1, 128, 0, stream>>>(out, gp_W1, gp_b1, gp_W2, gp_b2,
                                 out + (size_t)N * HDIM, N);
  k_mask<<<(N + 255) / 256, 256, 0, stream>>>(out + (size_t)N * HDIM + HDIM, N);
}
